// RoutedMoE_20925080666812
// MI455X (gfx1250) — compile-verified, matrix operand order checked by reading, not run on hardware
//
#include <hip/hip_runtime.h>
#include <hip/hip_bf16.h>

// ---------------------------------------------------------------------------
// MoE (Mixtral-style top-2) for gfx1250: grouped GEMM in bf16 WMMA.
// B tiles via Tensor Data Mover (TENSORcnt), gathered A via async->LDS
// (ASYNCcnt), double-buffered, K-step 64.
// ---------------------------------------------------------------------------

#define T_TOK   4096      // B*S
#define EMB_D   2048
#define MLP_D   4096
#define NEXP    8
#define TOPK    2
#define LDSK    72        // LDS row stride in bf16 elems (64 + 8 pad)

typedef __attribute__((ext_vector_type(16))) __bf16 bf16x16;
typedef __attribute__((ext_vector_type(8)))  float  floatx8;
typedef __attribute__((ext_vector_type(4)))  unsigned int uint32x4;
typedef __attribute__((ext_vector_type(8)))  int int32x8;
typedef __attribute__((ext_vector_type(4)))  int int32x4;

union FragAB { bf16x16 v; int4 q[2]; };

__device__ __forceinline__ floatx8 wmma_bf16(const FragAB& a, const FragAB& b, floatx8 c) {
    return __builtin_amdgcn_wmma_f32_16x16x32_bf16(
        /*neg_a=*/false, a.v, /*neg_b=*/false, b.v,
        /*c_mod=*/(short)0, c, /*reuse_a=*/false, /*reuse_b=*/false);
}

__device__ __forceinline__ unsigned f2bf(float f) {
    union { float f; unsigned u; } c; c.f = f;
    unsigned u = c.u;
    u += 0x7fffu + ((u >> 16) & 1u);   // round-to-nearest-even
    return u >> 16;
}

// CDNA5 async copy: global -> LDS, 16B per lane, tracked by ASYNCcnt.
__device__ __forceinline__ void async_ld_b128(const void* gsrc, void* ldst) {
    unsigned loff = (unsigned)(uintptr_t)ldst;
    asm volatile("global_load_async_to_lds_b128 %0, %1, off"
                 :: "v"(loff), "v"(gsrc)
                 : "memory");
}
#define WAIT_ASYNC(n) asm volatile("s_wait_asynccnt %0" :: "n"(n) : "memory")

// TDM: load a [128 rows x 64 bf16] tile from a row-major [tensorRows x rowLen]
// bf16 tensor into LDS, padding 4 DWORDs every 32 DWORDs (row stride = 72 bf16).
// This toolchain exposes the 6-arg builtin: (g0, g1, g2, g3, g4, cpol).
__device__ __forceinline__ void tdm_load_tile(const unsigned short* gbase,
                                              unsigned rowLenElems, unsigned tensorRows,
                                              void* lds) {
    unsigned long long ga = (unsigned long long)(uintptr_t)gbase;
    uint32x4 g0 = { 1u,                                     // count=1 (valid user D#)
                    (unsigned)(uintptr_t)lds,               // lds_addr
                    (unsigned)ga,                           // global_addr[31:0]
                    (unsigned)((ga >> 32) & 0x01FFFFFFu) | 0x80000000u }; // addr[56:32] | type=2
    int32x8 g1 = { (int)0x07110000,                         // data_size=2B, pad_en, intvl=32DW, amt=4DW
                   (int)((rowLenElems & 0xFFFFu) << 16),    // tensor_dim0 lo16 (<<48 overall)
                   (int)((rowLenElems >> 16) | ((tensorRows & 0xFFFFu) << 16)), // dim0 hi | dim1 lo
                   (int)((tensorRows >> 16) | (64u << 16)), // dim1 hi | tile_dim0 = 64
                   (int)128,                                // tile_dim1 = 128 rows
                   (int)rowLenElems,                        // tensor_dim0_stride lo32
                   0, 0 };
    int32x4 z4 = { 0, 0, 0, 0 };
    int32x8 z8 = { 0, 0, 0, 0, 0, 0, 0, 0 };
    __builtin_amdgcn_tensor_load_to_lds(g0, g1, z4, z4, z8, 0);
}

// ---------------------------------------------------------------------------
// 1) fp32 -> bf16 straight copy (for x)
// ---------------------------------------------------------------------------
__global__ void moe_cvt_bf16(const float* __restrict__ s, unsigned short* __restrict__ d, long n4) {
    long i  = (long)blockIdx.x * blockDim.x + threadIdx.x;
    long st = (long)gridDim.x * blockDim.x;
    const float4* s4 = (const float4*)s;
    uint2* d2 = (uint2*)d;
    for (; i < n4; i += st) {
        float4 v = s4[i];
        uint2 o;
        o.x = f2bf(v.x) | (f2bf(v.y) << 16);
        o.y = f2bf(v.z) | (f2bf(v.w) << 16);
        d2[i] = o;
    }
}

// ---------------------------------------------------------------------------
// 2) fp32 [batch][R][C] -> bf16 [batch][C][R] transpose-convert (weights)
// ---------------------------------------------------------------------------
__global__ void moe_transpose_bf16(const float* __restrict__ src, unsigned short* __restrict__ dst,
                                   int R, int C) {
    __shared__ float tile[32][33];
    const int b = blockIdx.z;
    const float* s = src + (size_t)b * R * C;
    unsigned short* d = dst + (size_t)b * R * C;
    const int c0 = blockIdx.x * 32, r0 = blockIdx.y * 32;
    const int x = threadIdx.x & 31, y4 = threadIdx.x >> 5;
#pragma unroll
    for (int i = 0; i < 4; i++) {
        int y = y4 * 4 + i;
        tile[y][x] = s[(size_t)(r0 + y) * C + c0 + x];
    }
    __syncthreads();
#pragma unroll
    for (int i = 0; i < 4; i++) {
        int y = y4 * 4 + i;
        d[(size_t)(c0 + y) * R + r0 + x] = (unsigned short)f2bf(tile[x][y]);
    }
}

// ---------------------------------------------------------------------------
// 3) Gate: one wave32 per token. logits -> top2 -> softmax -> counts
// ---------------------------------------------------------------------------
__global__ void moe_gate(const float* __restrict__ x, const float* __restrict__ gk,
                         int* __restrict__ counts, int* __restrict__ tok_expert,
                         float* __restrict__ tok_weight) {
    const int wave = threadIdx.x >> 5, lane = threadIdx.x & 31;
    const int t = blockIdx.x * 8 + wave;
    const float* xr = x + (size_t)t * EMB_D;
    float acc[NEXP];
#pragma unroll
    for (int e = 0; e < NEXP; e++) acc[e] = 0.f;
    for (int i = 0; i < EMB_D / 32; i++) {
        int d = i * 32 + lane;
        float xv = xr[d];
        const float4* g = (const float4*)(gk + (size_t)d * NEXP);
        float4 g0 = g[0], g1 = g[1];
        acc[0] += xv * g0.x; acc[1] += xv * g0.y; acc[2] += xv * g0.z; acc[3] += xv * g0.w;
        acc[4] += xv * g1.x; acc[5] += xv * g1.y; acc[6] += xv * g1.z; acc[7] += xv * g1.w;
    }
#pragma unroll
    for (int off = 16; off > 0; off >>= 1)
#pragma unroll
        for (int e = 0; e < NEXP; e++) acc[e] += __shfl_down(acc[e], off, 32);
    if (lane == 0) {
        float b0 = -3.0e38f; int i0 = 0;
#pragma unroll
        for (int e = 0; e < NEXP; e++) if (acc[e] > b0) { b0 = acc[e]; i0 = e; }
        float b1 = -3.0e38f; int i1 = 0;
#pragma unroll
        for (int e = 0; e < NEXP; e++) if (e != i0 && acc[e] > b1) { b1 = acc[e]; i1 = e; }
        float p1 = __expf(b1 - b0);
        float inv = __builtin_amdgcn_rcpf(1.0f + p1);
        tok_expert[2 * t + 0] = i0;  tok_weight[2 * t + 0] = inv;
        tok_expert[2 * t + 1] = i1;  tok_weight[2 * t + 1] = p1 * inv;
        atomicAdd(&counts[i0], 1);
        atomicAdd(&counts[i1], 1);
    }
}

// ---------------------------------------------------------------------------
// 4) tiny exclusive scan of the 8 counts
// ---------------------------------------------------------------------------
__global__ void moe_scan(const int* __restrict__ counts, int* __restrict__ offs) {
    if (threadIdx.x == 0 && blockIdx.x == 0) {
        int s = 0;
        for (int e = 0; e < NEXP; e++) { offs[e] = s; s += counts[e]; }
        offs[NEXP] = s;
    }
}

// ---------------------------------------------------------------------------
// 5) scatter tokens into compact per-expert slot arrays
// ---------------------------------------------------------------------------
__global__ void moe_scatter(const int* __restrict__ tok_expert, const float* __restrict__ tok_weight,
                            const int* __restrict__ offs, int* __restrict__ fill,
                            int* __restrict__ slot_token, float* __restrict__ slot_w) {
    int t = blockIdx.x * 256 + threadIdx.x;
    if (t >= T_TOK) return;
#pragma unroll
    for (int k = 0; k < TOPK; k++) {
        int e = tok_expert[2 * t + k];
        int pos = offs[e] + atomicAdd(&fill[e], 1);
        slot_token[pos] = t;
        slot_w[pos] = tok_weight[2 * t + k];
    }
}

// ---------------------------------------------------------------------------
// 6) GEMM1 (grouped, gathered A): h0 = Xe@W0e, h1 = Xe@W1e, inter = silu(h0)*h1
//    WG tile 64(M) x 128(N), K-step 64. B via TDM, A via async->LDS.
// ---------------------------------------------------------------------------
__global__ __launch_bounds__(256) void moe_gemm1(
    const unsigned short* __restrict__ xb,   // [T][EMB] bf16
    const unsigned short* __restrict__ wt0,  // [E][MLP][EMB] bf16 (transposed)
    const unsigned short* __restrict__ wt1,
    const int* __restrict__ offs,
    const int* __restrict__ slot_token,
    unsigned short* __restrict__ inter)      // [2T][MLP] bf16
{
    const int e = blockIdx.z;
    const int sBeg = offs[e];
    const int Ne = offs[e + 1] - sBeg;
    const int m0 = blockIdx.y * 64;
    if (m0 >= Ne) return;
    const int n0 = blockIdx.x * 128;

    __shared__ __align__(16) unsigned short As[2][64 * LDSK];
    __shared__ __align__(16) unsigned short Bs0[2][128 * LDSK];
    __shared__ __align__(16) unsigned short Bs1[2][128 * LDSK];
    __shared__ int rowTok[64];

    const int tid = threadIdx.x;
    if (tid < 64) {
        int r = m0 + tid;
        rowTok[tid] = slot_token[sBeg + (r < Ne ? r : Ne - 1)];
    }
    __syncthreads();

    const int wave = tid >> 5, lane = tid & 31;
    const int wm = wave & 3, wn = wave >> 2;
    const int khalf = lane >> 4, nm = lane & 15;

    floatx8 acc0[4], acc1[4];
    const floatx8 vz = {0.f, 0.f, 0.f, 0.f, 0.f, 0.f, 0.f, 0.f};
#pragma unroll
    for (int t = 0; t < 4; t++) { acc0[t] = vz; acc1[t] = vz; }

    const unsigned short* w0p = wt0 + (size_t)e * MLP_D * EMB_D;
    const unsigned short* w1p = wt1 + (size_t)e * MLP_D * EMB_D;

    // gathered A: 64 rows x 64 k per stage = 512 x b128 -> 2 per thread
    const int arow = tid >> 2;
    const int aseg0 = (tid & 3), aseg1 = (tid & 3) + 4;   // 8 segs of 8 bf16
    const long tokA = rowTok[arow];
    const unsigned short* gA = xb + tokA * EMB_D;

#define G1_ISSUE(st, k0)                                                        \
    do {                                                                        \
        async_ld_b128(gA + (k0) + aseg0 * 8, &As[st][arow * LDSK + aseg0 * 8]); \
        async_ld_b128(gA + (k0) + aseg1 * 8, &As[st][arow * LDSK + aseg1 * 8]); \
        if (wave == 0) {                                                        \
            tdm_load_tile(w0p + (size_t)n0 * EMB_D + (k0), EMB_D, MLP_D, &Bs0[st][0]); \
            tdm_load_tile(w1p + (size_t)n0 * EMB_D + (k0), EMB_D, MLP_D, &Bs1[st][0]); \
        }                                                                       \
    } while (0)

    constexpr int NSTEP = EMB_D / 64;
    G1_ISSUE(0, 0);

    for (int i = 0; i < NSTEP; i++) {
        const int cur = i & 1;
        if (i + 1 < NSTEP) {
            G1_ISSUE(cur ^ 1, (i + 1) * 64);
            WAIT_ASYNC(2);                                  // stage i's A loads done
            if (wave == 0) __builtin_amdgcn_s_wait_tensorcnt(2); // stage i's 2 TDMs done
        } else {
            WAIT_ASYNC(0);
            if (wave == 0) __builtin_amdgcn_s_wait_tensorcnt(0);
        }
        __syncthreads();          // all waves' stage-i data visible

#pragma unroll
        for (int ks = 0; ks < 2; ks++) {
            const int acol = ks * 32 + khalf * 8;
            FragAB af;
            af.q[0] = *(const int4*)&As[cur][(wm * 16 + nm) * LDSK + acol];
            af.q[1] = *(const int4*)&As[cur][(wm * 16 + nm) * LDSK + acol + 16];
#pragma unroll
            for (int t = 0; t < 4; t++) {
                int bn = wn * 64 + t * 16 + nm;
                FragAB b0f, b1f;
                b0f.q[0] = *(const int4*)&Bs0[cur][bn * LDSK + acol];
                b0f.q[1] = *(const int4*)&Bs0[cur][bn * LDSK + acol + 16];
                b1f.q[0] = *(const int4*)&Bs1[cur][bn * LDSK + acol];
                b1f.q[1] = *(const int4*)&Bs1[cur][bn * LDSK + acol + 16];
                acc0[t] = wmma_bf16(af, b0f, acc0[t]);
                acc1[t] = wmma_bf16(af, b1f, acc1[t]);
            }
        }
        __syncthreads();          // done reading buf[cur] before refill
    }
#undef G1_ISSUE

    // epilogue: silu(h0)*h1 -> bf16 inter
#pragma unroll
    for (int t = 0; t < 4; t++) {
        int col = n0 + wn * 64 + t * 16 + nm;
#pragma unroll
        for (int j = 0; j < 8; j++) {
            int r = m0 + wm * 16 + j + 8 * khalf;
            if (r < Ne) {
                float h0 = acc0[t][j], h1 = acc1[t][j];
                float sv = h0 * __builtin_amdgcn_rcpf(1.0f + __expf(-h0));
                inter[(size_t)(sBeg + r) * MLP_D + col] = (unsigned short)f2bf(sv * h1);
            }
        }
    }
}

// ---------------------------------------------------------------------------
// 7) GEMM2: out[token] += slot_w * (inter_slot @ Wo_e)   (fp32 atomics)
// ---------------------------------------------------------------------------
__global__ __launch_bounds__(256) void moe_gemm2(
    const unsigned short* __restrict__ inter, // [2T][MLP] bf16
    const unsigned short* __restrict__ wto,   // [E][EMB][MLP] bf16 (transposed)
    const int* __restrict__ offs,
    const int* __restrict__ slot_token,
    const float* __restrict__ slot_w,
    float* __restrict__ out)                  // [T][EMB] fp32 (zeroed)
{
    const int e = blockIdx.z;
    const int sBeg = offs[e];
    const int Ne = offs[e + 1] - sBeg;
    const int m0 = blockIdx.y * 64;
    if (m0 >= Ne) return;
    const int n0 = blockIdx.x * 128;

    __shared__ __align__(16) unsigned short As[2][64 * LDSK];
    __shared__ __align__(16) unsigned short Bs[2][128 * LDSK];

    const int tid = threadIdx.x;
    const int wave = tid >> 5, lane = tid & 31;
    const int wm = wave & 3, wn = wave >> 2;
    const int khalf = lane >> 4, nm = lane & 15;

    floatx8 acc[4];
    const floatx8 vz = {0.f, 0.f, 0.f, 0.f, 0.f, 0.f, 0.f, 0.f};
#pragma unroll
    for (int t = 0; t < 4; t++) acc[t] = vz;

    const unsigned short* wop = wto + (size_t)e * EMB_D * MLP_D;

    const int arow = tid >> 2;
    const int aseg0 = (tid & 3), aseg1 = (tid & 3) + 4;
    const int ar = m0 + arow;
    const long slotA = sBeg + (ar < Ne ? ar : Ne - 1);
    const unsigned short* gA = inter + slotA * MLP_D;

#define G2_ISSUE(st, k0)                                                        \
    do {                                                                        \
        async_ld_b128(gA + (k0) + aseg0 * 8, &As[st][arow * LDSK + aseg0 * 8]); \
        async_ld_b128(gA + (k0) + aseg1 * 8, &As[st][arow * LDSK + aseg1 * 8]); \
        if (wave == 0)                                                          \
            tdm_load_tile(wop + (size_t)n0 * MLP_D + (k0), MLP_D, EMB_D, &Bs[st][0]); \
    } while (0)

    constexpr int NSTEP = MLP_D / 64;
    G2_ISSUE(0, 0);

    for (int i = 0; i < NSTEP; i++) {
        const int cur = i & 1;
        if (i + 1 < NSTEP) {
            G2_ISSUE(cur ^ 1, (i + 1) * 64);
            WAIT_ASYNC(2);
            if (wave == 0) __builtin_amdgcn_s_wait_tensorcnt(1);
        } else {
            WAIT_ASYNC(0);
            if (wave == 0) __builtin_amdgcn_s_wait_tensorcnt(0);
        }
        __syncthreads();

#pragma unroll
        for (int ks = 0; ks < 2; ks++) {
            const int acol = ks * 32 + khalf * 8;
            FragAB af;
            af.q[0] = *(const int4*)&As[cur][(wm * 16 + nm) * LDSK + acol];
            af.q[1] = *(const int4*)&As[cur][(wm * 16 + nm) * LDSK + acol + 16];
#pragma unroll
            for (int t = 0; t < 4; t++) {
                int bn = wn * 64 + t * 16 + nm;
                FragAB bf;
                bf.q[0] = *(const int4*)&Bs[cur][bn * LDSK + acol];
                bf.q[1] = *(const int4*)&Bs[cur][bn * LDSK + acol + 16];
                acc[t] = wmma_bf16(af, bf, acc[t]);
            }
        }
        __syncthreads();
    }
#undef G2_ISSUE

#pragma unroll
    for (int t = 0; t < 4; t++) {
        int col = n0 + wn * 64 + t * 16 + nm;
#pragma unroll
        for (int j = 0; j < 8; j++) {
            int r = m0 + wm * 16 + j + 8 * khalf;
            if (r < Ne) {
                int slot = sBeg + r;
                float w = slot_w[slot];
                long tok = slot_token[slot];
                atomicAdd(out + tok * EMB_D + col, acc[t][j] * w);
            }
        }
    }
}

// ---------------------------------------------------------------------------
// Workspace layout (bytes)
// ---------------------------------------------------------------------------
static constexpr size_t XB_BYTES    = (size_t)T_TOK * EMB_D * 2;          // 16 MiB
static constexpr size_t WT_BYTES    = (size_t)NEXP * EMB_D * MLP_D * 2;   // 128 MiB each
static constexpr size_t INTER_BYTES = (size_t)T_TOK * TOPK * MLP_D * 2;   // 64 MiB
static constexpr size_t XB_OFF    = 0;
static constexpr size_t WT0_OFF   = XB_OFF + XB_BYTES;
static constexpr size_t WT1_OFF   = WT0_OFF + WT_BYTES;
static constexpr size_t WTO_OFF   = WT1_OFF + WT_BYTES;
static constexpr size_t INTER_OFF = WTO_OFF + WT_BYTES;
static constexpr size_t META_OFF  = INTER_OFF + INTER_BYTES;

extern "C" void kernel_launch(void* const* d_in, const int* in_sizes, int n_in,
                              void* d_out, int out_size, void* d_ws, size_t ws_size,
                              hipStream_t stream) {
    const float* x  = (const float*)d_in[0];
    const float* gk = (const float*)d_in[1];
    const float* w0 = (const float*)d_in[2];
    const float* w1 = (const float*)d_in[3];
    const float* wo = (const float*)d_in[4];
    float* out = (float*)d_out;

    char* ws = (char*)d_ws;
    unsigned short* xb    = (unsigned short*)(ws + XB_OFF);
    unsigned short* wt0   = (unsigned short*)(ws + WT0_OFF);
    unsigned short* wt1   = (unsigned short*)(ws + WT1_OFF);
    unsigned short* wto   = (unsigned short*)(ws + WTO_OFF);
    unsigned short* inter = (unsigned short*)(ws + INTER_OFF);
    int* meta = (int*)(ws + META_OFF);
    int*   counts     = meta + 0;
    int*   fill       = meta + 8;
    int*   offs       = meta + 16;               // 9 used, 16 reserved
    int*   tok_expert = meta + 32;               // 2T ints
    int*   slot_token = meta + 32 + 2 * T_TOK;   // 2T ints
    float* tok_weight = (float*)(meta + 32 + 4 * T_TOK);
    float* slot_w     = tok_weight + 2 * T_TOK;

    (void)hipMemsetAsync(meta, 0, 32 * sizeof(int), stream);                     // counts+fill
    (void)hipMemsetAsync(out, 0, (size_t)T_TOK * EMB_D * sizeof(float), stream); // atomic accum target

    // conversions
    moe_cvt_bf16<<<2048, 256, 0, stream>>>(x, xb, (long)T_TOK * EMB_D / 4);
    moe_transpose_bf16<<<dim3(MLP_D / 32, EMB_D / 32, NEXP), 256, 0, stream>>>(w0, wt0, EMB_D, MLP_D);
    moe_transpose_bf16<<<dim3(MLP_D / 32, EMB_D / 32, NEXP), 256, 0, stream>>>(w1, wt1, EMB_D, MLP_D);
    moe_transpose_bf16<<<dim3(EMB_D / 32, MLP_D / 32, NEXP), 256, 0, stream>>>(wo, wto, MLP_D, EMB_D);

    // routing
    moe_gate<<<T_TOK / 8, 256, 0, stream>>>(x, gk, counts, tok_expert, tok_weight);
    moe_scan<<<1, 32, 0, stream>>>(counts, offs);
    moe_scatter<<<T_TOK / 256, 256, 0, stream>>>(tok_expert, tok_weight, offs, fill, slot_token, slot_w);

    // grouped expert GEMMs
    moe_gemm1<<<dim3(MLP_D / 128, T_TOK / 64, NEXP), 256, 0, stream>>>(
        xb, wt0, wt1, offs, slot_token, inter);
    moe_gemm2<<<dim3(EMB_D / 128, T_TOK / 64, NEXP), 256, 0, stream>>>(
        inter, wto, offs, slot_token, slot_w, out);
}
